// HeadGatedCrossAttention_22428319220432
// MI455X (gfx1250) — matrix operand order repaired and checked by reading, not compiled, not run on hardware
//
#include <hip/hip_runtime.h>
#include <hip/hip_bf16.h>

// ---------------------------------------------------------------------------
// HeadGatedCrossAttention for MI455X (gfx1250, wave32, WMMA)
// B=16, LQ=576, LT=512, D=768, H=12, DH=64, TOPK=64
// All matmuls via v_wmma_f32_16x16x32_bf16; tile staging via
// global_load_async_to_lds_b128 (ASYNCcnt) where no transpose is needed.
// ---------------------------------------------------------------------------

#define BB   16
#define LQ   576
#define LT   512
#define DD   768
#define HH   12
#define DH   64
#define SCALE 0.125f   // DH^-0.5

typedef __attribute__((ext_vector_type(16))) __bf16 v16bf;
typedef __attribute__((ext_vector_type(8)))  float  v8f;

union AFrag {
    v16bf v;
    unsigned int u32[8];
    unsigned short u16[16];
};

static __device__ inline v8f v8zero() {
    v8f z = {0.f, 0.f, 0.f, 0.f, 0.f, 0.f, 0.f, 0.f};
    return z;
}

static __device__ inline unsigned short f32_to_bf16(float f) {
    unsigned int u = __float_as_uint(f);
    u += 0x7FFFu + ((u >> 16) & 1u);   // round-to-nearest-even
    return (unsigned short)(u >> 16);
}

static __device__ inline float bf16_to_f32(unsigned short u) {
    return __uint_as_float(((unsigned int)u) << 16);
}

// Async global -> LDS copy, 16 bytes per lane (GLOBAL_LOAD_ASYNC_TO_LDS_B128,
// GV addressing). LDS operand is the 32-bit truncation of the generic pointer
// (the aperture mapping hardware itself applies).
static __device__ inline void async_copy_b128(const void* gsrc, void* lds_dst) {
    unsigned int lds = (unsigned int)(uintptr_t)lds_dst;
    unsigned long long g = (unsigned long long)(uintptr_t)gsrc;
    asm volatile("global_load_async_to_lds_b128 %0, %1, off"
                 :: "v"(lds), "v"(g) : "memory");
}

static __device__ inline void async_wait0() {
    asm volatile("s_wait_asynccnt 0x0" ::: "memory");
}

// ---------------------------------------------------------------------------
// Kernel 1: LayerNorm (D=768) -> bf16.  One block (256 thr) per row.
// ---------------------------------------------------------------------------
__global__ __launch_bounds__(256) void ln_bf16_kernel(
    const float* __restrict__ x, const float* __restrict__ g,
    const float* __restrict__ b, unsigned short* __restrict__ y)
{
    __shared__ float wred[8];
    const int row = blockIdx.x;
    const int tid = threadIdx.x;
    const float* xr = x + (size_t)row * DD;

    float v0 = xr[tid], v1 = xr[tid + 256], v2 = xr[tid + 512];
    float s = v0 + v1 + v2;
    #pragma unroll
    for (int d = 16; d > 0; d >>= 1) s += __shfl_xor(s, d, 32);
    if ((tid & 31) == 0) wred[tid >> 5] = s;
    __syncthreads();
    float mean = 0.f;
    #pragma unroll
    for (int i = 0; i < 8; i++) mean += wred[i];
    mean *= (1.f / (float)DD);
    __syncthreads();

    float d0 = v0 - mean, d1 = v1 - mean, d2 = v2 - mean;
    float q = d0 * d0 + d1 * d1 + d2 * d2;
    #pragma unroll
    for (int d = 16; d > 0; d >>= 1) q += __shfl_xor(q, d, 32);
    if ((tid & 31) == 0) wred[tid >> 5] = q;
    __syncthreads();
    float var = 0.f;
    #pragma unroll
    for (int i = 0; i < 8; i++) var += wred[i];
    var *= (1.f / (float)DD);
    float rstd = rsqrtf(var + 1e-5f);

    unsigned short* yr = y + (size_t)row * DD;
    yr[tid]       = f32_to_bf16(d0 * rstd * g[tid]       + b[tid]);
    yr[tid + 256] = f32_to_bf16(d1 * rstd * g[tid + 256] + b[tid + 256]);
    yr[tid + 512] = f32_to_bf16(d2 * rstd * g[tid + 512] + b[tid + 512]);
}

// ---------------------------------------------------------------------------
// Kernel 2: fp32 -> bf16 bulk convert (weights)
// ---------------------------------------------------------------------------
__global__ __launch_bounds__(256) void cvt_bf16_kernel(
    const float* __restrict__ x, unsigned short* __restrict__ y, int n)
{
    int i = blockIdx.x * 256 + threadIdx.x;
    if (i < n) y[i] = f32_to_bf16(x[i]);
}

// ---------------------------------------------------------------------------
// Kernel 3: WMMA GEMM  C[m,n] = sum_k A[m,k] * W[n,k]   (torch Linear x@W.T)
// A bf16 [M,768], W bf16 [768,768].  Block tile: 128(M) x 64(N), 256 threads
// = 8 waves, each wave produces a 16x64 stripe (4 accumulators).
// Tiles staged with global_load_async_to_lds_b128 + s_wait_asynccnt.
// mode 0: write bf16 to [B,H,Lx,DH] layout (head split; grid.y == head)
// mode 1: write f32 d_out with fused residual: resid + sigmoid(logit)*C
// ---------------------------------------------------------------------------
__global__ __launch_bounds__(256) void gemm_bf16_wmma(
    const unsigned short* __restrict__ A, const unsigned short* __restrict__ W,
    int mode, unsigned short* __restrict__ out_bf16, int Lx,
    float* __restrict__ out_f32, const float* __restrict__ resid,
    const float* __restrict__ resid_logit)
{
    __shared__ unsigned short As[128 * 32];
    __shared__ unsigned short Ws[64 * 32];

    const int K    = DD;
    const int m0   = blockIdx.x * 128;
    const int n0   = blockIdx.y * 64;
    const int tid  = threadIdx.x;
    const int wave = tid >> 5;
    const int lane = tid & 31;
    const int am   = lane & 15;           // A row / B col within tile
    const int bn   = am;
    const int kb   = (lane >> 4) * 8;     // A K-base for this lane half
    const int kb2  = (lane >> 4) * 16;    // unused; B loads also use kb as K-base
    const int half8 = (lane >> 4) * 8;    // C row base

    v8f acc[4] = {v8zero(), v8zero(), v8zero(), v8zero()};

    for (int k0 = 0; k0 < K; k0 += 32) {
        if (k0 + 32 < K) {  // global_prefetch_b8 of the next K slab
            __builtin_prefetch(&A[(size_t)(m0 + (tid & 127)) * K + k0 + 32], 0, 1);
            if (tid < 64)
                __builtin_prefetch(&W[(size_t)(n0 + tid) * K + k0 + 32], 0, 1);
        }
        // A tile 128x32 bf16: 512 async B128 copies, 2 per thread
        {
            int r = tid >> 2, c = (tid & 3) << 3;
            async_copy_b128(&A[(size_t)(m0 + r) * K + k0 + c], &As[r * 32 + c]);
            int i2 = tid + 256;
            r = i2 >> 2; c = (i2 & 3) << 3;
            async_copy_b128(&A[(size_t)(m0 + r) * K + k0 + c], &As[r * 32 + c]);
        }
        // W tile 64x32 bf16: 256 async B128 copies, 1 per thread
        {
            int r = tid >> 2, c = (tid & 3) << 3;
            async_copy_b128(&W[(size_t)(n0 + r) * K + k0 + c], &Ws[r * 32 + c]);
        }
        async_wait0();
        __syncthreads();

        AFrag a;
        const unsigned short* ar = &As[(wave * 16 + am) * 32];
        #pragma unroll
        for (int j = 0; j < 4; j++) {
            a.u32[j]     = *(const unsigned int*)&ar[kb + 2 * j];
            a.u32[4 + j] = *(const unsigned int*)&ar[16 + kb + 2 * j];
        }
        #pragma unroll
        for (int c4 = 0; c4 < 4; c4++) {
            AFrag bf;
            const unsigned short* br = &Ws[(c4 * 16 + bn) * 32];
            #pragma unroll
            for (int j = 0; j < 8; j++)
                bf.u32[j] = *(const unsigned int*)&br[kb + ((j >> 2) << 4) + 2 * (j & 3)];
            acc[c4] = __builtin_amdgcn_wmma_f32_16x16x32_bf16(
                false, a.v, false, bf.v, (short)0, acc[c4], false, false);
        }
        __syncthreads();
    }

    if (mode == 0) {
        // n0 is a multiple of 64 and tile N == DH, so head == blockIdx.y
        #pragma unroll
        for (int c4 = 0; c4 < 4; c4++) {
            #pragma unroll
            for (int r = 0; r < 8; r++) {
                int m  = m0 + wave * 16 + half8 + r;
                int bb = m / Lx;
                int l  = m - bb * Lx;
                int dh = c4 * 16 + bn;
                out_bf16[(((size_t)bb * HH + blockIdx.y) * Lx + l) * DH + dh] =
                    f32_to_bf16(acc[c4][r]);
            }
        }
    } else {
        float alpha = 1.f / (1.f + __expf(-resid_logit[0]));
        #pragma unroll
        for (int c4 = 0; c4 < 4; c4++) {
            #pragma unroll
            for (int r = 0; r < 8; r++) {
                int m = m0 + wave * 16 + half8 + r;
                int n = n0 + c4 * 16 + bn;
                size_t idx = (size_t)m * DD + n;
                out_f32[idx] = resid[idx] + alpha * acc[c4][r];
            }
        }
    }
}

// ---------------------------------------------------------------------------
// Kernel 4: per-(b,h) gate: scores = hq.K * scale, softmax, c_h, sigmoid
// gate; bitonic sort (LDS) for top-64 threshold -> keep mask.
// ---------------------------------------------------------------------------
__global__ __launch_bounds__(256) void gate_topk_kernel(
    const unsigned short* __restrict__ Kb, const float* __restrict__ hq,
    const int* __restrict__ mask, float* __restrict__ g_h,
    unsigned char* __restrict__ keep)
{
    __shared__ float s[LT];
    __shared__ float ssort[LT];
    __shared__ float cpart[4][DH];
    __shared__ float wred[8];
    __shared__ float hqs[DH];
    __shared__ float red2[DH];

    const int bh  = blockIdx.x;
    const int b   = bh / HH;
    const int h   = bh - b * HH;
    const int tid = threadIdx.x;
    const unsigned short* Kp = Kb + (size_t)bh * LT * DH;

    if (tid < DH) hqs[tid] = hq[h * DH + tid];
    __syncthreads();

    // scores
    for (int t = tid; t < LT; t += 256) {
        float accd = 0.f;
        const unsigned short* kr = Kp + (size_t)t * DH;
        #pragma unroll
        for (int d = 0; d < DH; d++) accd += bf16_to_f32(kr[d]) * hqs[d];
        accd *= SCALE;
        if (mask[b * LT + t] == 0) accd = -1e30f;
        s[t] = accd;
    }
    __syncthreads();

    // block max over 512
    float v = fmaxf(s[tid], s[tid + 256]);
    #pragma unroll
    for (int d = 16; d > 0; d >>= 1) v = fmaxf(v, __shfl_xor(v, d, 32));
    if ((tid & 31) == 0) wred[tid >> 5] = v;
    __syncthreads();
    float mx = wred[0];
    #pragma unroll
    for (int i = 1; i < 8; i++) mx = fmaxf(mx, wred[i]);
    __syncthreads();

    // p = exp(s - mx); keep order-preserving copy for top-k sort
    float p0 = __expf(s[tid] - mx);
    float p1 = __expf(s[tid + 256] - mx);
    s[tid] = p0;           s[tid + 256] = p1;
    ssort[tid] = p0;       ssort[tid + 256] = p1;
    float sv = p0 + p1;
    #pragma unroll
    for (int d = 16; d > 0; d >>= 1) sv += __shfl_xor(sv, d, 32);
    if ((tid & 31) == 0) wred[tid >> 5] = sv;
    __syncthreads();
    float denom = 0.f;
    #pragma unroll
    for (int i = 0; i < 8; i++) denom += wred[i];
    denom = fmaxf(denom, 1e-20f);
    __syncthreads();

    // bitonic sort ascending (512 elems, 256 threads, 2 pairs/thread)
    for (int k2 = 2; k2 <= LT; k2 <<= 1) {
        for (int j = k2 >> 1; j > 0; j >>= 1) {
            for (int i = tid; i < LT; i += 256) {
                int ixj = i ^ j;
                if (ixj > i) {
                    bool up = ((i & k2) == 0);
                    float aa = ssort[i], cc = ssort[ixj];
                    if ((aa > cc) == up) { ssort[i] = cc; ssort[ixj] = aa; }
                }
            }
            __syncthreads();
        }
    }
    float thr = ssort[LT - 64];   // 64th largest (exp is monotonic)
    keep[(size_t)bh * LT + tid]       = (s[tid]       >= thr) ? 1 : 0;
    keep[(size_t)bh * LT + tid + 256] = (s[tid + 256] >= thr) ? 1 : 0;

    // c_h = sum_l attn[l] * K[l,:]  (4-way split over tokens)
    const int d   = tid & 63;
    const int grp = tid >> 6;
    float accc = 0.f;
    for (int l = grp; l < LT; l += 4)
        accc += s[l] * bf16_to_f32(Kp[(size_t)l * DH + d]);
    cpart[grp][d] = accc;
    __syncthreads();
    if (tid < DH) {
        float ch = (cpart[0][tid] + cpart[1][tid] + cpart[2][tid] + cpart[3][tid]) / denom;
        red2[tid] = ch * hqs[tid];
    }
    __syncthreads();
    if (tid == 0) {
        float gl = 0.f;
        #pragma unroll
        for (int i = 0; i < DH; i++) gl += red2[i];
        gl *= SCALE;
        g_h[bh] = 1.f / (1.f + __expf(-gl));
    }
}

// ---------------------------------------------------------------------------
// Kernel 5: zero out K/V for non-top-k tokens (matches torch.where -> 0)
// ---------------------------------------------------------------------------
__global__ __launch_bounds__(256) void apply_topk_kernel(
    unsigned short* __restrict__ Kb, unsigned short* __restrict__ Vb,
    const unsigned char* __restrict__ keep)
{
    size_t i   = (size_t)blockIdx.x * 256 + threadIdx.x;
    size_t tok = i >> 6;
    if (!keep[tok]) { Kb[i] = 0; Vb[i] = 0; }
}

// ---------------------------------------------------------------------------
// Kernel 6: flash attention per (b,h,qtile=64).  128 threads = 4 waves,
// each wave owns 16 q rows.  KV tiles of 32 tokens. All matmuls via WMMA.
// K tile staged via async-to-LDS; V transposed through registers.
// ---------------------------------------------------------------------------
__global__ __launch_bounds__(128) void attn_wmma_kernel(
    const unsigned short* __restrict__ Qb, const unsigned short* __restrict__ Kb,
    const unsigned short* __restrict__ Vb, const float* __restrict__ g_h,
    const int* __restrict__ mask, unsigned short* __restrict__ Ob)
{
    __shared__ unsigned short Ks[32 * 64];     // [token][dh]
    __shared__ unsigned short VTs[64 * 32];    // [dh][token]
    __shared__ unsigned short Ps[4][16 * 32];  // per-wave P staging

    const int blk  = blockIdx.x;
    const int qt   = blk % (LQ / 64);
    const int bh   = blk / (LQ / 64);
    const int b    = bh / HH;
    const int h    = bh - b * HH;
    const int tid  = threadIdx.x;
    const int wave = tid >> 5;
    const int lane = tid & 31;
    const int am   = lane & 15;
    const int bn   = am;
    const int kb   = (lane >> 4) * 8;
    const int kb2  = (lane >> 4) * 16;
    const int half8 = (lane >> 4) * 8;

    const unsigned short* Qp = Qb + ((size_t)bh * LQ + qt * 64 + wave * 16) * DH;
    const unsigned short* Kp = Kb + (size_t)bh * LT * DH;
    const unsigned short* Vp = Vb + (size_t)bh * LT * DH;
    const int* mp = mask + b * LT;

    // Q fragments for K = dh 0..31 and 32..63 (held across all KV tiles)
    AFrag qa0, qa1;
    {
        const unsigned short* qr = Qp + am * DH;
        #pragma unroll
        for (int j = 0; j < 4; j++) {
            qa0.u32[j]     = *(const unsigned int*)&qr[kb + 2 * j];
            qa0.u32[4 + j] = *(const unsigned int*)&qr[16 + kb + 2 * j];
            qa1.u32[j]     = *(const unsigned int*)&qr[32 + kb + 2 * j];
            qa1.u32[4 + j] = *(const unsigned int*)&qr[48 + kb + 2 * j];
        }
    }

    v8f o[4] = {v8zero(), v8zero(), v8zero(), v8zero()};
    float mrow[8], lrow[8];
    #pragma unroll
    for (int r = 0; r < 8; r++) { mrow[r] = -1e30f; lrow[r] = 0.f; }

    for (int kt = 0; kt < LT / 32; kt++) {
        // K tile [32][64] via async-to-LDS; V tile transposed via registers
        #pragma unroll
        for (int i = tid; i < 256; i += 128) {
            int r = i >> 3, c = (i & 7) << 3;
            async_copy_b128(&Kp[((size_t)kt * 32 + r) * DH + c], &Ks[r * 64 + c]);
            uint4 vv = *(const uint4*)&Vp[((size_t)kt * 32 + r) * DH + c];
            const unsigned short* ve = (const unsigned short*)&vv;
            #pragma unroll
            for (int j = 0; j < 8; j++) VTs[(c + j) * 32 + r] = ve[j];
        }
        async_wait0();
        __syncthreads();

        // S = Q K^T  (two 16-token column sub-tiles, contraction dh=64)
        v8f sS[2];
        #pragma unroll
        for (int sub = 0; sub < 2; sub++) {
            v8f t = v8zero();
            const unsigned short* kr = &Ks[(sub * 16 + bn) * 64];
            AFrag bf;
            #pragma unroll
            for (int j = 0; j < 8; j++)
                bf.u32[j] = *(const unsigned int*)&kr[kb + ((j >> 2) << 4) + 2 * (j & 3)];
            t = __builtin_amdgcn_wmma_f32_16x16x32_bf16(
                false, qa0.v, false, bf.v, (short)0, t, false, false);
            #pragma unroll
            for (int j = 0; j < 8; j++)
                bf.u32[j] = *(const unsigned int*)&kr[32 + kb + ((j >> 2) << 4) + 2 * (j & 3)];
            t = __builtin_amdgcn_wmma_f32_16x16x32_bf16(
                false, qa1.v, false, bf.v, (short)0, t, false, false);
            int tok = kt * 32 + sub * 16 + bn;
            bool ok = (mp[tok] != 0);
            #pragma unroll
            for (int e = 0; e < 8; e++) t[e] = ok ? t[e] * SCALE : -1e30f;
            sS[sub] = t;
        }

        // online softmax per row (rows live in VGPR idx + lane half;
        // columns are lanes within the half -> shfl_xor masks 1,2,4,8)
        #pragma unroll
        for (int r = 0; r < 8; r++) {
            float cur = fmaxf(sS[0][r], sS[1][r]);
            #pragma unroll
            for (int d2 = 1; d2 < 16; d2 <<= 1)
                cur = fmaxf(cur, __shfl_xor(cur, d2, 32));
            float mnew = fmaxf(mrow[r], cur);
            float corr = __expf(mrow[r] - mnew);
            float pa = __expf(sS[0][r] - mnew);
            float pb = __expf(sS[1][r] - mnew);
            sS[0][r] = pa; sS[1][r] = pb;
            float rs = pa + pb;
            #pragma unroll
            for (int d2 = 1; d2 < 16; d2 <<= 1)
                rs += __shfl_xor(rs, d2, 32);
            lrow[r] = lrow[r] * corr + rs;
            mrow[r] = mnew;
            #pragma unroll
            for (int c4 = 0; c4 < 4; c4++) o[c4][r] *= corr;
        }

        // stage P (C layout -> row-major 16x32 bf16) in per-wave LDS;
        // same-wave DS ordering makes the immediate re-load safe.
        #pragma unroll
        for (int r = 0; r < 8; r++) {
            Ps[wave][(half8 + r) * 32 + bn]      = f32_to_bf16(sS[0][r]);
            Ps[wave][(half8 + r) * 32 + 16 + bn] = f32_to_bf16(sS[1][r]);
        }

        // O += P V   (A = P 16x32, B = V 32xdh via transposed tile)
        AFrag pa2;
        const unsigned short* pr = &Ps[wave][am * 32];
        #pragma unroll
        for (int j = 0; j < 4; j++) {
            pa2.u32[j]     = *(const unsigned int*)&pr[kb + 2 * j];
            pa2.u32[4 + j] = *(const unsigned int*)&pr[16 + kb + 2 * j];
        }
        #pragma unroll
        for (int c4 = 0; c4 < 4; c4++) {
            AFrag vf;
            const unsigned short* vr = &VTs[(c4 * 16 + bn) * 32];
            #pragma unroll
            for (int j = 0; j < 8; j++)
                vf.u32[j] = *(const unsigned int*)&vr[kb + ((j >> 2) << 4) + 2 * (j & 3)];
            o[c4] = __builtin_amdgcn_wmma_f32_16x16x32_bf16(
                false, pa2.v, false, vf.v, (short)0, o[c4], false, false);
        }
        __syncthreads();
    }

    // epilogue: 1/l, head gate, write [B, LQ, H*DH] bf16
    float gh = g_h[bh];
    #pragma unroll
    for (int c4 = 0; c4 < 4; c4++) {
        #pragma unroll
        for (int r = 0; r < 8; r++) {
            int lq = qt * 64 + wave * 16 + half8 + r;
            float val = o[c4][r] / fmaxf(lrow[r], 1e-20f) * gh;
            Ob[((size_t)b * LQ + lq) * DD + h * DH + c4 * 16 + bn] =
                f32_to_bf16(val);
        }
    }
}

// ---------------------------------------------------------------------------
// Host launcher
// ---------------------------------------------------------------------------
extern "C" void kernel_launch(void* const* d_in, const int* in_sizes, int n_in,
                              void* d_out, int out_size, void* d_ws, size_t ws_size,
                              hipStream_t stream) {
    const float* x_q    = (const float*)d_in[0];
    const float* x_kv   = (const float*)d_in[1];
    const int*   kvmask = (const int*)  d_in[2];
    const float* wq     = (const float*)d_in[3];
    const float* wk     = (const float*)d_in[4];
    const float* wv     = (const float*)d_in[5];
    const float* wo     = (const float*)d_in[6];
    const float* lnqg   = (const float*)d_in[7];
    const float* lnqb   = (const float*)d_in[8];
    const float* lnkvg  = (const float*)d_in[9];
    const float* lnkvb  = (const float*)d_in[10];
    const float* hq     = (const float*)d_in[11];
    const float* rlogit = (const float*)d_in[12];

    char* ws = (char*)d_ws;
    size_t off = 0;
    auto alloc = [&](size_t bytes) -> void* {
        void* p = ws + off;
        off += (bytes + 255) & ~(size_t)255;
        return p;
    };

    const size_t MQ = (size_t)BB * LQ;   // 9216
    const size_t MK = (size_t)BB * LT;   // 8192

    unsigned short* xqn  = (unsigned short*)alloc(MQ * DD * 2);
    unsigned short* xkvn = (unsigned short*)alloc(MK * DD * 2);
    unsigned short* wqb  = (unsigned short*)alloc((size_t)DD * DD * 2);
    unsigned short* wkb  = (unsigned short*)alloc((size_t)DD * DD * 2);
    unsigned short* wvb  = (unsigned short*)alloc((size_t)DD * DD * 2);
    unsigned short* wob  = (unsigned short*)alloc((size_t)DD * DD * 2);
    unsigned short* Qbuf = (unsigned short*)alloc(MQ * DD * 2);
    unsigned short* Kbuf = (unsigned short*)alloc(MK * DD * 2);
    unsigned short* Vbuf = (unsigned short*)alloc(MK * DD * 2);
    unsigned short* Obuf = (unsigned short*)alloc(MQ * DD * 2);
    float*          ghb  = (float*)alloc((size_t)BB * HH * 4);
    unsigned char*  keep = (unsigned char*)alloc((size_t)BB * HH * LT);

    // 1) layernorm -> bf16
    ln_bf16_kernel<<<(int)MQ, 256, 0, stream>>>(x_q,  lnqg,  lnqb,  xqn);
    ln_bf16_kernel<<<(int)MK, 256, 0, stream>>>(x_kv, lnkvg, lnkvb, xkvn);

    // 2) weights -> bf16
    const int WN = DD * DD;
    cvt_bf16_kernel<<<(WN + 255) / 256, 256, 0, stream>>>(wq, wqb, WN);
    cvt_bf16_kernel<<<(WN + 255) / 256, 256, 0, stream>>>(wk, wkb, WN);
    cvt_bf16_kernel<<<(WN + 255) / 256, 256, 0, stream>>>(wv, wvb, WN);
    cvt_bf16_kernel<<<(WN + 255) / 256, 256, 0, stream>>>(wo, wob, WN);

    // 3) QKV projections (WMMA), direct to [B,H,L,DH] bf16
    gemm_bf16_wmma<<<dim3((unsigned)(MQ / 128), HH), 256, 0, stream>>>(
        xqn, wqb, 0, Qbuf, LQ, nullptr, nullptr, nullptr);
    gemm_bf16_wmma<<<dim3((unsigned)(MK / 128), HH), 256, 0, stream>>>(
        xkvn, wkb, 0, Kbuf, LT, nullptr, nullptr, nullptr);
    gemm_bf16_wmma<<<dim3((unsigned)(MK / 128), HH), 256, 0, stream>>>(
        xkvn, wvb, 0, Vbuf, LT, nullptr, nullptr, nullptr);

    // 4) head gates + top-k keep mask (uses pre-pruned K, as in reference)
    gate_topk_kernel<<<BB * HH, 256, 0, stream>>>(Kbuf, hq, kvmask, ghb, keep);

    // 5) zero pruned K/V
    {
        size_t total = (size_t)BB * HH * LT * DH;
        apply_topk_kernel<<<(unsigned)(total / 256), 256, 0, stream>>>(
            Kbuf, Vbuf, keep);
    }

    // 6) flash attention (WMMA), gated epilogue -> [B,LQ,H*DH] bf16
    attn_wmma_kernel<<<BB * HH * (LQ / 64), 128, 0, stream>>>(
        Qbuf, Kbuf, Vbuf, ghb, kvmask, Obuf);

    // 7) output projection + residual -> f32 d_out
    gemm_bf16_wmma<<<dim3((unsigned)(MQ / 128), HH), 256, 0, stream>>>(
        Obuf, wob, 1, nullptr, LQ, (float*)d_out, x_q, rlogit);
}
